// HeARTMTHybrid_25005299597898
// MI455X (gfx1250) — compile-verified, matrix operand order checked by reading, not run on hardware
//
#include <hip/hip_runtime.h>
#include <cmath>

// ---------------------------------------------------------------------------
// Model constants
// ---------------------------------------------------------------------------
constexpr int kB    = 16;          // batch
constexpr int kL    = 197;         // sequence (1 + 14*14)
constexpr int kD    = 768;         // d_model
constexpr int kFF   = 3072;        // d_ff
constexpr int kDI   = 1536;        // d_inner (mamba)
constexpr int kST   = 16;          // d_state
constexpr int kDTR  = 48;          // dt_rank
constexpr int kH    = 12;          // heads
constexpr int kHD   = 64;          // head dim
constexpr int kTok  = kB * kL;     // 3152 token rows
constexpr int kP    = 196;         // patches per image
constexpr int kPTok = kB * kP;     // 3136 patch rows

typedef __attribute__((ext_vector_type(16))) __bf16 v16bf;
typedef __attribute__((ext_vector_type(8)))  float  v8f;

// LDS tile pitch: 32 live bf16 + 8 pad (16B) to spread fragment rows across banks
constexpr int kPitch = 40;

// ---------------------------------------------------------------------------
// Tensor Data Mover (CDNA5): build a 2D D# and issue tensor_load_to_lds.
// Probe-verified builtin; arity differs by toolchain (5-arg ROCm7.2, 6-arg
// therock which also ships the TDM header -> use __has_include to pick).
// ---------------------------------------------------------------------------
#if __has_builtin(__builtin_amdgcn_tensor_load_to_lds)
#define HAS_TDM 1
#else
#define HAS_TDM 0
#endif

#if HAS_TDM
typedef __attribute__((ext_vector_type(4))) unsigned int tdm_v4u;
typedef __attribute__((ext_vector_type(8))) int          tdm_v8i;
typedef __attribute__((ext_vector_type(4))) int          tdm_v4i;

// D# group1 d0 extras: pad_enable, pad_interval=16 DWORD (code 3), pad_amount=4 DWORD (code 3)
// -> after each 64B row written to LDS, skip 16B: reproduces the kPitch=40 row pitch.
#define TDM_PAD_BITS ((1u << 20) | (3u << 22) | (3u << 25))

__device__ __forceinline__ void tdm_load_2d(unsigned lds_off, unsigned long long gaddr,
                                            unsigned krun, unsigned rows,
                                            unsigned long long rstride_elems,
                                            unsigned flags) {
  tdm_v4u g0;
  g0[0] = 1u;                                        // count=1, user-mode descriptor
  g0[1] = lds_off;                                   // LDS byte address
  g0[2] = (unsigned)gaddr;                           // global_addr[31:0]
  g0[3] = (unsigned)(gaddr >> 32) | (2u << 30);      // global_addr[56:32] | type=2
  tdm_v8i g1;
  g1[0] = (int)((1u << 16) | flags);                 // wg_mask=0, data_size=1 (2B), pad cfg
  g1[1] = (int)((krun & 0xFFFFu) << 16);             // tensor_dim0[15:0]
  g1[2] = (int)(((krun >> 16) & 0xFFFFu) |
                ((rows & 0xFFFFu) << 16));           // tensor_dim0[31:16] | tensor_dim1[15:0]
  g1[3] = (int)(((rows >> 16) & 0xFFFFu) |
                ((krun & 0xFFFFu) << 16));           // tensor_dim1[31:16] | tile_dim0
  g1[4] = (int)(rows & 0xFFFFu);                     // tile_dim1 (tile_dim2 = 0)
  g1[5] = (int)(unsigned)(rstride_elems & 0xFFFFFFFFull);       // dim0_stride lo
  g1[6] = (int)(unsigned)((rstride_elems >> 32) & 0xFFFFull);   // dim0_stride hi
  g1[7] = 0;
  tdm_v4i gz = {};
#if __has_include(<hip/amd_detail/amd_gfx1250_TDM.h>)
  tdm_v8i gz8 = {};
  __builtin_amdgcn_tensor_load_to_lds(g0, g1, gz, gz, gz8, 0);
#else
  __builtin_amdgcn_tensor_load_to_lds(g0, g1, gz, gz, 0);
#endif
}
#endif  // HAS_TDM

// ---------------------------------------------------------------------------
// Generic batched-strided bf16 WMMA GEMM:  C = A(MxK) * B(KxN)  (+bias)(+act)(+resid)
// Block tile 128x128, BK=32 (one v_wmma_f32_16x16x32_bf16 K-step), 8 waves (2x4),
// wave tile 64x32 = 4x2 WMMA accumulators, double-buffered LDS.
// Both sA and sB are stored fragment-major [row/col][k] so every fragment load
// is two contiguous 16B LDS reads (ds_load_b128), no VALU repacking.
// A tiles stream through the Tensor Data Mover when fully in-bounds.
// ---------------------------------------------------------------------------
struct GemmArgs {
  const __bf16* A; long a_rs, a_cs, a_b0, a_b1;
  const __bf16* B; long b_rs, b_cs, b_b0, b_b1;
  float* C; __bf16* Cbf; long c_rs, c_b0, c_b1;   // C col-stride == 1
  const float* bias;        // indexed by n, may be null
  const float* resid;       // same layout as C, may be null
  int M, N, K, nb1, ep;     // ep: 0 none, 1 gelu(exact), 2 softplus
};

__global__ __launch_bounds__(256)
void gemm_kernel(GemmArgs g) {
  __shared__ __bf16 sA[2][128][kPitch];
  __shared__ __bf16 sB[2][128][kPitch];   // transposed: [n][k]

  const int tid   = threadIdx.x;
  const int wave  = tid >> 5;
  const int lane  = tid & 31;
  const int klane = lane >> 4;       // 0: K-low half, 1: K-high half (16-bit layout)
  const int lrow  = lane & 15;
  const int wm    = wave >> 2;       // 0..1
  const int wn    = wave & 3;        // 0..3

  const int zb = (int)(blockIdx.z) / g.nb1;
  const int zh = (int)(blockIdx.z) % g.nb1;
  const __bf16* Ab = g.A + (long)zb * g.a_b0 + (long)zh * g.a_b1;
  const __bf16* Bb = g.B + (long)zb * g.b_b0 + (long)zh * g.b_b1;
  const long cbase = (long)zb * g.c_b0 + (long)zh * g.c_b1;

  const int m0 = blockIdx.y * 128;
  const int n0 = blockIdx.x * 128;

  v8f acc[4][2] = {};

  const int kTiles = (g.K + 31) >> 5;

  auto stage = [&](int kt, int buf) {
    const int kbase = kt << 5;
#if HAS_TDM
    // ---- A tile via Tensor Data Mover (unit K-stride, fully in-bounds) ----
    if (g.a_cs == 1 && (m0 + 128 <= g.M) && (kbase + 32 <= g.K)) {
      if (tid == 0) {
        tdm_load_2d((unsigned)(unsigned long long)&sA[buf][0][0],
                    (unsigned long long)(const void*)(Ab + (long)m0 * g.a_rs + kbase),
                    32u, 128u, (unsigned long long)g.a_rs, TDM_PAD_BITS);
      }
    } else
#endif
    {
      for (int i = tid; i < 128 * 32; i += 256) {
        const int r = i >> 5, c = i & 31;
        const int m = m0 + r, k = kbase + c;
        __bf16 v = (__bf16)0.0f;
        if (m < g.M && k < g.K) v = Ab[(long)m * g.a_rs + (long)k * g.a_cs];
        sA[buf][r][c] = v;
      }
    }
    // ---- B tile: global [k][n] (coalesced along n) -> LDS transposed [n][k] ----
    for (int i = tid; i < 32 * 128; i += 256) {
      const int r = i >> 7, c = i & 127;       // r = k offset, c = n offset
      const int k = kbase + r, n = n0 + c;
      __bf16 v = (__bf16)0.0f;
      if (k < g.K && n < g.N) v = Bb[(long)k * g.b_rs + (long)n * g.b_cs];
      sB[buf][c][r] = v;
    }
    // keep the next-next tiles warming in L2 (global_prefetch_b8)
    if ((tid & 63) == 0 && kbase + 64 < g.K) {
      __builtin_prefetch(Ab + (long)m0 * g.a_rs + (long)(kbase + 64) * g.a_cs, 0, 1);
      __builtin_prefetch(Bb + (long)(kbase + 64) * g.b_rs + (long)n0 * g.b_cs, 0, 1);
    }
  };

  stage(0, 0);
#if HAS_TDM && __has_builtin(__builtin_amdgcn_s_wait_tensorcnt)
  __builtin_amdgcn_s_wait_tensorcnt(0);
#endif
  __syncthreads();

  for (int kt = 0; kt < kTiles; ++kt) {
    const int buf = kt & 1;
    if (kt + 1 < kTiles) stage(kt + 1, buf ^ 1);   // TDM/loads overlap the WMMAs below

    v16bf afrag[4];
    v16bf bfrag[2];
#pragma unroll
    for (int e = 0; e < 16; ++e) {
      // 16-bit fragment layout (ISA 7.12.2): elem e -> K = (e&7) + (e>>3)*16 + klane*8
      const int k = (e & 7) + ((e >> 3) << 4) + (klane << 3);
#pragma unroll
      for (int fm = 0; fm < 4; ++fm)
        afrag[fm][e] = sA[buf][wm * 64 + fm * 16 + lrow][k];
#pragma unroll
      for (int fn = 0; fn < 2; ++fn)
        bfrag[fn][e] = sB[buf][wn * 32 + fn * 16 + lrow][k];
    }

#pragma unroll
    for (int fm = 0; fm < 4; ++fm)
#pragma unroll
      for (int fn = 0; fn < 2; ++fn)
        acc[fm][fn] = __builtin_amdgcn_wmma_f32_16x16x32_bf16(
            false, afrag[fm], false, bfrag[fn],
            (short)0, acc[fm][fn], false, false);

#if HAS_TDM && __has_builtin(__builtin_amdgcn_s_wait_tensorcnt)
    __builtin_amdgcn_s_wait_tensorcnt(0);
#endif
    __syncthreads();
  }

  // Epilogue: C/D fragment layout: VGPR i -> row (i + 8*klane), col lrow
#pragma unroll
  for (int fm = 0; fm < 4; ++fm) {
#pragma unroll
    for (int fn = 0; fn < 2; ++fn) {
      const int n = n0 + wn * 32 + fn * 16 + lrow;
      if (n >= g.N) continue;
      const int mb = m0 + wm * 64 + fm * 16 + klane * 8;
#pragma unroll
      for (int i = 0; i < 8; ++i) {
        const int m = mb + i;
        if (m >= g.M) continue;
        float v = acc[fm][fn][i];
        if (g.bias) v += g.bias[n];
        if (g.ep == 1) {                         // exact GELU
          v = 0.5f * v * (1.0f + erff(v * 0.70710678118f));
        } else if (g.ep == 2) {                  // softplus
          v = (v > 20.0f) ? v : log1pf(__expf(v));
        }
        const long idx = cbase + (long)m * g.c_rs + (long)n;
        if (g.resid) v += g.resid[idx];
        if (g.C)   g.C[idx]   = v;
        if (g.Cbf) g.Cbf[idx] = (__bf16)v;
      }
    }
  }
}

// ---------------------------------------------------------------------------
// fp32 -> bf16 weight conversion (optional transpose: src[r][c] -> dst[c][r])
// ---------------------------------------------------------------------------
__global__ void cvt_kernel(const float* src, __bf16* dst, long rows, long cols, int trans) {
  const long i = (long)blockIdx.x * 256 + threadIdx.x;
  const long n = rows * cols;
  if (i >= n) return;
  if (trans) {
    const long r = i / cols, c = i % cols;
    dst[c * rows + r] = (__bf16)src[i];
  } else {
    dst[i] = (__bf16)src[i];
  }
}

// ---------------------------------------------------------------------------
// LayerNorm over last dim D; one block per row; optional fp32 and bf16 outputs
// ---------------------------------------------------------------------------
__global__ __launch_bounds__(256)
void ln_kernel(const float* x, const float* w, const float* b,
               float* y, __bf16* ybf, int D, float eps) {
  __shared__ float r1[256];
  __shared__ float r2[256];
  const long row = blockIdx.x;
  const float* xr = x + row * (long)D;
  const int t = threadIdx.x;
  float s = 0.f, ss = 0.f;
  for (int j = t; j < D; j += 256) { const float v = xr[j]; s += v; ss += v * v; }
  r1[t] = s; r2[t] = ss;
  __syncthreads();
  for (int o = 128; o > 0; o >>= 1) {
    if (t < o) { r1[t] += r1[t + o]; r2[t] += r2[t + o]; }
    __syncthreads();
  }
  const float mean = r1[0] / (float)D;
  const float var  = r2[0] / (float)D - mean * mean;
  const float inv  = rsqrtf(var + eps);
  for (int j = t; j < D; j += 256) {
    const float v = (xr[j] - mean) * inv * w[j] + b[j];
    if (y)   y[row * (long)D + j]   = v;
    if (ybf) ybf[row * (long)D + j] = (__bf16)v;
  }
}

// ---------------------------------------------------------------------------
// Softmax over rows of S (length L), scale folded in, bf16 output
// ---------------------------------------------------------------------------
__global__ __launch_bounds__(256)
void softmax_kernel(const float* S, __bf16* P, int L, float scale) {
  __shared__ float red[256];
  const long row = blockIdx.x;
  const float* s = S + row * (long)L;
  const int t = threadIdx.x;
  float m = -1e30f;
  for (int j = t; j < L; j += 256) m = fmaxf(m, s[j] * scale);
  red[t] = m; __syncthreads();
  for (int o = 128; o > 0; o >>= 1) { if (t < o) red[t] = fmaxf(red[t], red[t + o]); __syncthreads(); }
  m = red[0];
  __syncthreads();
  float sum = 0.f;
  for (int j = t; j < L; j += 256) sum += __expf(s[j] * scale - m);
  red[t] = sum; __syncthreads();
  for (int o = 128; o > 0; o >>= 1) { if (t < o) red[t] += red[t + o]; __syncthreads(); }
  const float invs = 1.0f / red[0];
  for (int j = t; j < L; j += 256)
    P[row * (long)L + j] = (__bf16)(__expf(s[j] * scale - m) * invs);
}

// ---------------------------------------------------------------------------
// Patchify: NCHW image -> bf16 im2col rows [(b,patch), (c,py,px)]  (K = 768)
// ---------------------------------------------------------------------------
__global__ void im2col_kernel(const float* img, __bf16* out) {
  const long i = (long)blockIdx.x * 256 + threadIdx.x;
  const long n = (long)kPTok * kD;
  if (i >= n) return;
  const int k  = (int)(i % kD);
  const long t = i / kD;
  const int gp = (int)(t % kP);
  const int b  = (int)(t / kP);
  const int gx = gp % 14, gy = gp / 14;
  const int px = k % 16, py = (k / 16) % 16, c = k / 256;
  out[i] = (__bf16)img[(((long)b * 3 + c) * 224 + (gy * 16 + py)) * 224 + (gx * 16 + px)];
}

// x[b,0] = cls + pos[0];  x[b,1+p] = ptmp[b,p] + pos[1+p]
__global__ void embed_kernel(const float* ptmp, const float* cls, const float* pos, float* x) {
  const long i = (long)blockIdx.x * 256 + threadIdx.x;
  const long n = (long)kTok * kD;
  if (i >= n) return;
  const int d  = (int)(i % kD);
  const long t = i / kD;
  const int l  = (int)(t % kL);
  const int b  = (int)(t / kL);
  const float v = (l == 0) ? cls[d] : ptmp[((long)b * kP + (l - 1)) * kD + d];
  x[i] = v + pos[(long)l * kD + d];
}

// ---------------------------------------------------------------------------
// Mamba depthwise causal conv (D_CONV=4) + SiLU; xm lives in xz[..., :kDI]
// ---------------------------------------------------------------------------
__global__ void conv_kernel(const float* xz, const float* cw, const float* cb,
                            float* xc, __bf16* xcbf) {
  const long i = (long)blockIdx.x * 256 + threadIdx.x;
  const long n = (long)kTok * kDI;
  if (i >= n) return;
  const int c  = (int)(i % kDI);
  const long t = i / kDI;
  const int l  = (int)(t % kL);
  const long b = t / kL;
  float a = cb[c];
#pragma unroll
  for (int k = 0; k < 4; ++k) {
    const int ls = l + k - 3;
    if (ls >= 0) a += cw[(long)c * 4 + k] * xz[((long)b * kL + ls) * (2 * kDI) + c];
  }
  const float s = a / (1.0f + __expf(-a));   // silu
  xc[i]   = s;
  xcbf[i] = (__bf16)s;
}

// ---------------------------------------------------------------------------
// Selective scan: one thread per (batch, channel); 16-state recurrence over L,
// fused with +D*xc and SiLU(z) gating; emits bf16 for the out-projection GEMM.
// ---------------------------------------------------------------------------
__global__ void scan_kernel(const float* dtb, const float* dbl, const float* A_log,
                            const float* Dmv, const float* xc, const float* xz,
                            __bf16* ybf) {
  const long idx = (long)blockIdx.x * 256 + threadIdx.x;
  if (idx >= (long)kB * kDI) return;
  const int b = (int)(idx / kDI);
  const int c = (int)(idx % kDI);
  float A[kST];
  float h[kST];
#pragma unroll
  for (int s = 0; s < kST; ++s) { A[s] = -__expf(A_log[(long)c * kST + s]); h[s] = 0.f; }
  const float Dc = Dmv[c];
  for (int l = 0; l < kL; ++l) {
    const long t = (long)b * kL + l;
    const float dtv = dtb[t * kDI + c];
    const float xcv = xc[t * kDI + c];
    const float* bl = dbl + t * (kDTR + 2 * kST) + kDTR;   // B at [48:64), C at [64:80)
    float y = 0.f;
#pragma unroll
    for (int s = 0; s < kST; ++s) {
      h[s] = __expf(dtv * A[s]) * h[s] + dtv * bl[s] * xcv;
      y += h[s] * bl[kST + s];
    }
    y += Dc * xcv;
    const float z = xz[t * (2 * kDI) + kDI + c];
    y *= z / (1.0f + __expf(-z));            // * silu(z)
    ybf[t * kDI + c] = (__bf16)y;
  }
}

// Flip bf16 activations along the sequence axis
__global__ void flip_bf_kernel(const __bf16* in, __bf16* out) {
  const long i = (long)blockIdx.x * 256 + threadIdx.x;
  const long n = (long)kTok * kD;
  if (i >= n) return;
  const int d  = (int)(i % kD);
  const long t = i / kD;
  const int l  = (int)(t % kL);
  const long b = t / kL;
  out[i] = in[((long)b * kL + (kL - 1 - l)) * kD + d];
}

// x += of + flip(ob)
__global__ void add3_kernel(float* x, const float* of, const float* ob) {
  const long i = (long)blockIdx.x * 256 + threadIdx.x;
  const long n = (long)kTok * kD;
  if (i >= n) return;
  const int d  = (int)(i % kD);
  const long t = i / kD;
  const int l  = (int)(t % kL);
  const long b = t / kL;
  x[i] += of[i] + ob[((long)b * kL + (kL - 1 - l)) * kD + d];
}

// ---------------------------------------------------------------------------
// Host orchestration
// ---------------------------------------------------------------------------
extern "C" void kernel_launch(void* const* d_in, const int* in_sizes, int n_in,
                              void* d_out, int out_size, void* d_ws, size_t ws_size,
                              hipStream_t stream) {
  (void)in_sizes; (void)n_in; (void)out_size; (void)ws_size;

  const float* pixel   = (const float*)d_in[0];
  const float* patch_w = (const float*)d_in[1];
  const float* patch_b = (const float*)d_in[2];
  const float* cls_tok = (const float*)d_in[3];
  const float* pos_emb = (const float*)d_in[4];
  const float* ln1_w   = (const float*)d_in[5];
  const float* ln1_b   = (const float*)d_in[6];
  const float* qkv_w   = (const float*)d_in[7];
  const float* qkv_b   = (const float*)d_in[8];
  const float* ao_w    = (const float*)d_in[9];
  const float* ao_b    = (const float*)d_in[10];
  const float* ln2_w   = (const float*)d_in[11];
  const float* ln2_b   = (const float*)d_in[12];
  const float* mlp1_w  = (const float*)d_in[13];
  const float* mlp1_b  = (const float*)d_in[14];
  const float* mlp2_w  = (const float*)d_in[15];
  const float* mlp2_b  = (const float*)d_in[16];
  const float* mnorm_w = (const float*)d_in[17];
  const float* mnorm_b = (const float*)d_in[18];
  const float* in_w    = (const float*)d_in[19];
  const float* conv_w  = (const float*)d_in[20];
  const float* conv_b  = (const float*)d_in[21];
  const float* xproj_w = (const float*)d_in[22];
  const float* dt_w    = (const float*)d_in[23];
  const float* dt_b    = (const float*)d_in[24];
  const float* A_log   = (const float*)d_in[25];
  const float* Dmv     = (const float*)d_in[26];
  const float* out_w   = (const float*)d_in[27];
  const float* fin_w   = (const float*)d_in[28];
  const float* fin_b   = (const float*)d_in[29];
  float* out = (float*)d_out;

  // --- workspace carve (bump allocator, 256B aligned) ---
  char* wp = (char*)d_ws;
  auto carve = [&](size_t bytes) -> void* {
    void* r = (void*)wp;
    wp += (bytes + 255) & ~(size_t)255;
    return r;
  };
  // per-layer-reused bf16 weight buffers
  __bf16* wqkv = (__bf16*)carve((size_t)kD * 3 * kD * 2);
  __bf16* wao  = (__bf16*)carve((size_t)kD * kD * 2);
  __bf16* wm1  = (__bf16*)carve((size_t)kD * kFF * 2);
  __bf16* wm2  = (__bf16*)carve((size_t)kFF * kD * 2);
  __bf16* wpT  = (__bf16*)carve((size_t)kD * kD * 2);
  __bf16* win  = (__bf16*)carve((size_t)kD * 2 * kDI * 2);
  __bf16* wxp  = (__bf16*)carve((size_t)kDI * (kDTR + 2 * kST) * 2);
  __bf16* wdt  = (__bf16*)carve((size_t)kDTR * kDI * 2);
  __bf16* wout = (__bf16*)carve((size_t)kDI * kD * 2);
  // activations
  __bf16* patches = (__bf16*)carve((size_t)kPTok * kD * 2);
  float*  ptmp    = (float*) carve((size_t)kPTok * kD * 4);
  float*  x       = (float*) carve((size_t)kTok * kD * 4);
  __bf16* lnbf    = (__bf16*)carve((size_t)kTok * kD * 2);
  __bf16* qkvbf   = (__bf16*)carve((size_t)kTok * 3 * kD * 2);
  float*  Smat    = (float*) carve((size_t)kB * kH * kL * kL * 4);
  __bf16* Pmat    = (__bf16*)carve((size_t)kB * kH * kL * kL * 2);
  __bf16* aobf    = (__bf16*)carve((size_t)kTok * kD * 2);
  __bf16* h2bf    = (__bf16*)carve((size_t)kTok * kFF * 2);
  __bf16* xnflip  = (__bf16*)carve((size_t)kTok * kD * 2);
  float*  xz      = (float*) carve((size_t)kTok * 2 * kDI * 4);
  float*  xc      = (float*) carve((size_t)kTok * kDI * 4);
  __bf16* xcbf    = (__bf16*)carve((size_t)kTok * kDI * 2);
  float*  dbl     = (float*) carve((size_t)kTok * (kDTR + 2 * kST) * 4);
  __bf16* dblbf   = (__bf16*)carve((size_t)kTok * (kDTR + 2 * kST) * 2);
  float*  dtb     = (float*) carve((size_t)kTok * kDI * 4);
  __bf16* ybf     = (__bf16*)carve((size_t)kTok * kDI * 2);
  float*  of      = (float*) carve((size_t)kTok * kD * 4);
  float*  ob      = (float*) carve((size_t)kTok * kD * 4);

  auto blocks1d = [](long n) { return dim3((unsigned)((n + 255) / 256)); };

  auto cvt = [&](const float* s, __bf16* dst, long rows, long cols, int tr) {
    cvt_kernel<<<blocks1d(rows * cols), 256, 0, stream>>>(s, dst, rows, cols, tr);
  };

  auto gemm = [&](const __bf16* A, long ars, long acs, long ab0, long ab1,
                  const __bf16* Bw, long brs, long bcs, long bb0, long bb1,
                  float* C, __bf16* Cbf, long crs, long cb0, long cb1,
                  const float* bias, const float* resid,
                  int M, int N, int K, int nb0, int nb1, int ep) {
    GemmArgs g{A, ars, acs, ab0, ab1,
               Bw, brs, bcs, bb0, bb1,
               C, Cbf, crs, cb0, cb1,
               bias, resid, M, N, K, nb1, ep};
    dim3 grid((unsigned)((N + 127) / 128), (unsigned)((M + 127) / 128), (unsigned)(nb0 * nb1));
    gemm_kernel<<<grid, dim3(256), 0, stream>>>(g);
  };

  // ---------------- patch embedding ----------------
  cvt(patch_w, wpT, kD, kD, /*transpose*/1);            // [O][3*16*16] -> [K][O]
  im2col_kernel<<<blocks1d((long)kPTok * kD), 256, 0, stream>>>(pixel, patches);
  gemm(patches, kD, 1, 0, 0,
       wpT, kD, 1, 0, 0,
       ptmp, nullptr, kD, 0, 0,
       patch_b, nullptr, kPTok, kD, kD, 1, 1, 0);
  embed_kernel<<<blocks1d((long)kTok * kD), 256, 0, stream>>>(ptmp, cls_tok, pos_emb, x);

  // ---------------- transformer layer ----------------
  auto tlayer = [&](int i) {
    cvt(qkv_w + (long)i * kD * 3 * kD, wqkv, kD, 3 * kD, 0);
    cvt(ao_w  + (long)i * kD * kD,     wao,  kD, kD, 0);
    cvt(mlp1_w+ (long)i * kD * kFF,    wm1,  kD, kFF, 0);
    cvt(mlp2_w+ (long)i * kFF * kD,    wm2,  kFF, kD, 0);

    ln_kernel<<<kTok, 256, 0, stream>>>(x, ln1_w + (long)i * kD, ln1_b + (long)i * kD,
                                        nullptr, lnbf, kD, 1e-12f);
    // QKV (bf16 mirror only)
    gemm(lnbf, kD, 1, 0, 0,
         wqkv, 3 * kD, 1, 0, 0,
         nullptr, qkvbf, 3 * kD, 0, 0,
         qkv_b + (long)i * 3 * kD, nullptr, kTok, 3 * kD, kD, 1, 1, 0);
    // scores: S[b,h] = Q K^T   (192 batched 197x197x64)
    gemm(qkvbf,            3 * kD, 1, (long)kL * 3 * kD, kHD,           // Q
         qkvbf + kD,       1, 3 * kD, (long)kL * 3 * kD, kHD,           // K^T
         Smat, nullptr, kL, (long)kH * kL * kL, (long)kL * kL,
         nullptr, nullptr, kL, kL, kHD, kB, kH, 0);
    softmax_kernel<<<kB * kH * kL, 256, 0, stream>>>(Smat, Pmat, kL, 0.125f);
    // O = P V  -> bf16 [b, l, h*64+d]
    gemm(Pmat, kL, 1, (long)kH * kL * kL, (long)kL * kL,
         qkvbf + 2 * kD, 3 * kD, 1, (long)kL * 3 * kD, kHD,
         nullptr, aobf, kD, (long)kL * kD, kHD,
         nullptr, nullptr, kL, kHD, kL, kB, kH, 0);
    // proj + residual
    gemm(aobf, kD, 1, 0, 0,
         wao, kD, 1, 0, 0,
         x, nullptr, kD, 0, 0,
         ao_b + (long)i * kD, x, kTok, kD, kD, 1, 1, 0);
    // MLP
    ln_kernel<<<kTok, 256, 0, stream>>>(x, ln2_w + (long)i * kD, ln2_b + (long)i * kD,
                                        nullptr, lnbf, kD, 1e-12f);
    gemm(lnbf, kD, 1, 0, 0,
         wm1, kFF, 1, 0, 0,
         nullptr, h2bf, kFF, 0, 0,
         mlp1_b + (long)i * kFF, nullptr, kTok, kFF, kD, 1, 1, /*gelu*/1);
    gemm(h2bf, kFF, 1, 0, 0,
         wm2, kD, 1, 0, 0,
         x, nullptr, kD, 0, 0,
         mlp2_b + (long)i * kD, x, kTok, kD, kFF, 1, 1, 0);
  };

  // ---------------- mamba module ----------------
  auto mamba = [&](int mi, const __bf16* xin, float* outf) {
    cvt(in_w    + (long)mi * kD * 2 * kDI,             win,  kD, 2 * kDI, 0);
    cvt(xproj_w + (long)mi * kDI * (kDTR + 2 * kST),   wxp,  kDI, kDTR + 2 * kST, 0);
    cvt(dt_w    + (long)mi * kDTR * kDI,               wdt,  kDTR, kDI, 0);
    cvt(out_w   + (long)mi * kDI * kD,                 wout, kDI, kD, 0);

    gemm(xin, kD, 1, 0, 0,
         win, 2 * kDI, 1, 0, 0,
         xz, nullptr, 2 * kDI, 0, 0,
         nullptr, nullptr, kTok, 2 * kDI, kD, 1, 1, 0);
    conv_kernel<<<blocks1d((long)kTok * kDI), 256, 0, stream>>>(
        xz, conv_w + (long)mi * kDI * 4, conv_b + (long)mi * kDI, xc, xcbf);
    gemm(xcbf, kDI, 1, 0, 0,
         wxp, kDTR + 2 * kST, 1, 0, 0,
         dbl, dblbf, kDTR + 2 * kST, 0, 0,
         nullptr, nullptr, kTok, kDTR + 2 * kST, kDI, 1, 1, 0);
    // dt = softplus(dbl[:, :48] @ dt_w + dt_b)
    gemm(dblbf, kDTR + 2 * kST, 1, 0, 0,
         wdt, kDI, 1, 0, 0,
         dtb, nullptr, kDI, 0, 0,
         dt_b + (long)mi * kDI, nullptr, kTok, kDI, kDTR, 1, 1, /*softplus*/2);
    scan_kernel<<<blocks1d((long)kB * kDI), 256, 0, stream>>>(
        dtb, dbl, A_log + (long)mi * kDI * kST, Dmv + (long)mi * kDI, xc, xz, ybf);
    gemm(ybf, kDI, 1, 0, 0,
         wout, kD, 1, 0, 0,
         outf, nullptr, kD, 0, 0,
         nullptr, nullptr, kTok, kD, kDI, 1, 1, 0);
  };

  // ---------------- schedule ----------------
  for (int i = 0; i < 8; ++i) tlayer(i);

  for (int j = 0; j < 4; ++j) {
    ln_kernel<<<kTok, 256, 0, stream>>>(x, mnorm_w + (long)j * kD, mnorm_b + (long)j * kD,
                                        nullptr, lnbf, kD, 1e-5f);
    mamba(2 * j, lnbf, of);                                        // forward
    flip_bf_kernel<<<blocks1d((long)kTok * kD), 256, 0, stream>>>(lnbf, xnflip);
    mamba(2 * j + 1, xnflip, ob);                                  // backward (flipped)
    add3_kernel<<<blocks1d((long)kTok * kD), 256, 0, stream>>>(x, of, ob);
  }

  for (int i = 8; i < 16; ++i) tlayer(i);

  // final LN straight into d_out (fp32)
  ln_kernel<<<kTok, 256, 0, stream>>>(x, fin_w, fin_b, out, nullptr, kD, 1e-12f);
}
